// MaskedMultiHeadAttention_46651934769354
// MI455X (gfx1250) — matrix operand order repaired and checked by reading, not compiled, not run on hardware
//
#include <hip/hip_runtime.h>

typedef __attribute__((ext_vector_type(16))) __bf16 v16bf;
typedef __attribute__((ext_vector_type(8)))  __bf16 v8bf;
typedef __attribute__((ext_vector_type(8)))  float  v8f;
typedef __attribute__((ext_vector_type(4)))  float  v4f;

#define B_ 4
#define S_ 2048
#define D_ 1024
#define H_ 16
#define HD_ 64

static __device__ __forceinline__ v8f wmma_bf16(v16bf a, v16bf b, v8f c) {
  return __builtin_amdgcn_wmma_f32_16x16x32_bf16(false, a, false, b, (short)0, c,
                                                 false, false);
}

static __device__ __forceinline__ v16bf cat8(v8bf lo, v8bf hi) {
  v16bf r;
#pragma unroll
  for (int i = 0; i < 8; ++i) { r[i] = lo[i]; r[8 + i] = hi[i]; }
  return r;
}

// Convert 16 fp32 (4x v4f) -> v16bf
static __device__ __forceinline__ v16bf cvt16(v4f f0, v4f f1, v4f f2, v4f f3) {
  v16bf r;
#pragma unroll
  for (int i = 0; i < 4; ++i) {
    r[i]      = (__bf16)f0[i];
    r[4 + i]  = (__bf16)f1[i];
    r[8 + i]  = (__bf16)f2[i];
    r[12 + i] = (__bf16)f3[i];
  }
  return r;
}

// Butterfly exchange across lanes via ds_swizzle (group-of-32 XOR mode):
// offset = {1'b0, xor_mask[4:0], or_mask[4:0]=0, and_mask[4:0]=31}
template <int MASK>
static __device__ __forceinline__ float xswz(float v) {
  return __int_as_float(
      __builtin_amdgcn_ds_swizzle(__float_as_int(v), (MASK << 10) | 31));
}

// max across the 16-lane half holding one C-layout row
static __device__ __forceinline__ float halfmax(float v) {
  v = fmaxf(v, xswz<1>(v));
  v = fmaxf(v, xswz<2>(v));
  v = fmaxf(v, xswz<4>(v));
  v = fmaxf(v, xswz<8>(v));
  return v;
}

// ---------------------------------------------------------------------------
// GEMM: C[16x16 tile] = A[M,K] * W[N,K]^T + bias
// MODE 0: A fp32, out = (C+b)*0.125 -> bf16 Q buffer [B,H,S,HD]
// MODE 1: A fp32, out = (C+b)       -> bf16 K buffer [B,H,S,HD]
// MODE 2: A fp32, out = (C+b)       -> bf16 V^T buffer [B,H,HD,S]
// MODE 3: A bf16, out = (C+b)       -> fp32 row-major [M,N]
// One wave per output tile. K = 1024. M = 8192, N = 1024.
// ---------------------------------------------------------------------------
template <int MODE>
__global__ __launch_bounds__(128) void gemm16(const float* __restrict__ Af,
                                              const __bf16* __restrict__ Ab,
                                              const float* __restrict__ W,
                                              const float* __restrict__ bias,
                                              __bf16* __restrict__ outb,
                                              float* __restrict__ outf) {
  constexpr int K = 1024;
  const int ln     = threadIdx.x & 31;
  const int wv     = threadIdx.x >> 5;
  const int half   = ln >> 4;      // which 16-lane half
  const int lane16 = ln & 15;
  const int m0 = blockIdx.x * 16;
  const int n0 = (blockIdx.y * 4 + wv) * 16;

  const int arow  = m0 + lane16;       // A-operand: row per lane
  const int abase = half * 8;          // K sub-chunk base (A layout)
  const int bn    = n0 + lane16;       // B-operand: column per lane
  const int bkoff = half * 8;          // K sub-chunk base (B layout)

  v8f acc = {};

  for (int k0 = 0; k0 < K; k0 += 32) {
    v16bf a;
    if constexpr (MODE < 3) {
      const float* ap = Af + (long)arow * K + k0 + abase;
      v4f f0 = *(const v4f*)(ap + 0);
      v4f f1 = *(const v4f*)(ap + 4);
      v4f f2 = *(const v4f*)(ap + 16);
      v4f f3 = *(const v4f*)(ap + 20);
      a = cvt16(f0, f1, f2, f3);
    } else {
      const __bf16* ap = Ab + (long)arow * K + k0 + abase;
      v8bf c0 = *(const v8bf*)(ap + 0);
      v8bf c1 = *(const v8bf*)(ap + 16);
      a = cat8(c0, c1);
    }
    const float* wp = W + (long)bn * K + k0 + bkoff;
    v4f w0 = *(const v4f*)(wp + 0);
    v4f w1 = *(const v4f*)(wp + 4);
    v4f w2 = *(const v4f*)(wp + 16);
    v4f w3 = *(const v4f*)(wp + 20);
    v16bf b = cvt16(w0, w1, w2, w3);

    acc = wmma_bf16(a, b, acc);
  }

  const float bv = bias[bn];

#pragma unroll
  for (int r = 0; r < 8; ++r) {
    const int m = m0 + r + half * 8;      // C layout: row = r + 8*(lane/16)
    float v = acc[r] + bv;
    if constexpr (MODE == 0) v *= 0.125f; // fold 1/sqrt(HD)
    if constexpr (MODE == 0 || MODE == 1) {
      const int b_ = m >> 11, s_ = m & 2047;
      const int h_ = bn >> 6, hd = bn & 63;
      outb[(((long)(b_ * H_ + h_) * S_) + s_) * HD_ + hd] = (__bf16)v;
    } else if constexpr (MODE == 2) {
      const int b_ = m >> 11, s_ = m & 2047;
      const int h_ = bn >> 6, hd = bn & 63;
      outb[(((long)(b_ * H_ + h_) * HD_) + hd) * S_ + s_] = (__bf16)v;
    } else {
      outf[(long)m * D_ + bn] = v;
    }
  }
}

// ---------------------------------------------------------------------------
// Causal flash attention. Q,K: [B*H, S, HD] bf16 (Q pre-scaled). Vt: [B*H, HD, S].
// Block = 4 waves; wave w owns 32 q rows (two stacked 16-row WMMA sub-tiles),
// so every K/V B-operand load feeds 2 WMMAs. Key tiles of 32.
// Row-sum of P is computed with a WMMA against a ones matrix (result already
// replicated per-row in C layout); row-max uses ds_swizzle XOR butterflies.
// ---------------------------------------------------------------------------
__global__ __launch_bounds__(128) void attn_kernel(const __bf16* __restrict__ Qb,
                                                   const __bf16* __restrict__ Kb,
                                                   const __bf16* __restrict__ Vt,
                                                   __bf16* __restrict__ aout) {
  __shared__ __bf16 lds[4][32 * 32];

  const int bh = blockIdx.y;
  const int b_ = bh >> 4;
  const int h_ = bh & 15;
  const int wv = threadIdx.x >> 5;
  const int ln = threadIdx.x & 31;
  const int half = ln >> 4;
  const int lane16 = ln & 15;
  const int qt0 = blockIdx.x * 128 + wv * 32;

  // Q A-operands: [subtile u][d-step t], each 16x32 over head dim
  v16bf qa[2][2];
#pragma unroll
  for (int u = 0; u < 2; ++u) {
    const __bf16* qp = Qb + ((long)bh * S_ + qt0 + u * 16 + lane16) * HD_;
#pragma unroll
    for (int t = 0; t < 2; ++t) {
      v8bf c0 = *(const v8bf*)(qp + t * 32 + half * 8);
      v8bf c1 = *(const v8bf*)(qp + t * 32 + half * 8 + 16);
      qa[u][t] = cat8(c0, c1);
    }
  }

  v16bf ones;
#pragma unroll
  for (int i = 0; i < 16; ++i) ones[i] = (__bf16)1.0f;

  v8f o[2][4] = {{{}, {}, {}, {}}, {{}, {}, {}, {}}};
  float mrow[2][8], lrow[2][8];
#pragma unroll
  for (int u = 0; u < 2; ++u)
#pragma unroll
    for (int r = 0; r < 8; ++r) { mrow[u][r] = -__builtin_inff(); lrow[u][r] = 0.0f; }

  for (int j0 = 0; j0 < qt0 + 32; j0 += 32) {
    // ---- scores: 32 q x 32 keys; each K operand feeds both sub-tiles ----
    v8f s[2][2] = {{{}, {}}, {{}, {}}};  // [u][nt]
#pragma unroll
    for (int t = 0; t < 2; ++t) {
#pragma unroll
      for (int nt = 0; nt < 2; ++nt) {
        const __bf16* kp =
            Kb + ((long)bh * S_ + j0 + nt * 16 + lane16) * HD_ + t * 32 + half * 8;
        v16bf kb = cat8(*(const v8bf*)(kp), *(const v8bf*)(kp + 16));
        s[0][nt] = wmma_bf16(qa[0][t], kb, s[0][nt]);
        s[1][nt] = wmma_bf16(qa[1][t], kb, s[1][nt]);
      }
    }

    const bool needmask = (j0 + 31 > qt0);

    // ---- online softmax; row lives in one 16-lane half ----
#pragma unroll
    for (int u = 0; u < 2; ++u) {
#pragma unroll
      for (int r = 0; r < 8; ++r) {
        const int row = qt0 + u * 16 + r + half * 8;
        if (needmask) {
          if (j0 + lane16 > row)      s[u][0][r] = -__builtin_inff();
          if (j0 + 16 + lane16 > row) s[u][1][r] = -__builtin_inff();
        }
        const float tmax = halfmax(fmaxf(s[u][0][r], s[u][1][r]));
        const float mnew = fmaxf(mrow[u][r], tmax);
        const float corr = __expf(mrow[u][r] - mnew);
        mrow[u][r] = mnew;
        lrow[u][r] *= corr;
#pragma unroll
        for (int nt = 0; nt < 4; ++nt) o[u][nt][r] *= corr;
        const float p0 = __expf(s[u][0][r] - mnew);
        const float p1 = __expf(s[u][1][r] - mnew);
        const int pr = u * 16 + r + half * 8;  // C-layout row within 32x32 P
        lds[wv][pr * 32 + lane16]      = (__bf16)p0;
        lds[wv][pr * 32 + 16 + lane16] = (__bf16)p1;
      }
    }

    // wave-local LDS round trip: re-lay P (C layout) into A-operand layout
    asm volatile("s_wait_dscnt 0x0" ::: "memory");
    v16bf pa[2];
#pragma unroll
    for (int u = 0; u < 2; ++u) {
      const __bf16* pp = &lds[wv][(u * 16 + lane16) * 32 + half * 8];
      pa[u] = cat8(*(const v8bf*)(pp), *(const v8bf*)(pp + 16));
    }

    // ---- row-sum of P via WMMA against ones (replicated across columns) ----
#pragma unroll
    for (int u = 0; u < 2; ++u) {
      v8f zero = {};
      v8f st = wmma_bf16(pa[u], ones, zero);
#pragma unroll
      for (int r = 0; r < 8; ++r) lrow[u][r] += st[r];
    }

    // ---- O += P(16x32) * V(32x64): 4 n-tiles; each V operand feeds 2 WMMAs ----
#pragma unroll
    for (int nt = 0; nt < 4; ++nt) {
      const __bf16* vp =
          Vt + ((long)bh * HD_ + nt * 16 + lane16) * S_ + j0 + half * 8;
      v16bf vb = cat8(*(const v8bf*)(vp), *(const v8bf*)(vp + 16));
      o[0][nt] = wmma_bf16(pa[0], vb, o[0][nt]);
      o[1][nt] = wmma_bf16(pa[1], vb, o[1][nt]);
    }
  }

  // ---- epilogue: normalize + scatter to [B,S,D] bf16 ----
#pragma unroll
  for (int u = 0; u < 2; ++u) {
#pragma unroll
    for (int r = 0; r < 8; ++r) {
      const int srow = qt0 + u * 16 + r + half * 8;
      const float inv = 1.0f / lrow[u][r];
      const long base = ((long)b_ * S_ + srow) * D_ + h_ * HD_;
#pragma unroll
      for (int nt = 0; nt < 4; ++nt)
        aout[base + nt * 16 + lane16] = (__bf16)(o[u][nt][r] * inv);
    }
  }
}

// ---------------------------------------------------------------------------
// Host launcher
// ---------------------------------------------------------------------------
extern "C" void kernel_launch(void* const* d_in, const int* in_sizes, int n_in,
                              void* d_out, int out_size, void* d_ws, size_t ws_size,
                              hipStream_t stream) {
  const float* q_in = (const float*)d_in[0];
  const float* k_in = (const float*)d_in[1];
  const float* v_in = (const float*)d_in[2];
  // d_in[3] = mask (causal; implicit in kernel)
  const float* wq = (const float*)d_in[4];
  const float* bq = (const float*)d_in[5];
  const float* wk = (const float*)d_in[6];
  const float* bk = (const float*)d_in[7];
  const float* wv = (const float*)d_in[8];
  const float* bv = (const float*)d_in[9];
  const float* wo = (const float*)d_in[10];
  const float* bo = (const float*)d_in[11];
  float* out = (float*)d_out;

  const size_t seg = (size_t)B_ * S_ * D_ * sizeof(__bf16);  // 16 MB
  char* ws = (char*)d_ws;
  __bf16* Qb  = (__bf16*)(ws + 0 * seg);
  __bf16* Kb  = (__bf16*)(ws + 1 * seg);
  __bf16* Vt  = (__bf16*)(ws + 2 * seg);
  __bf16* Att = (__bf16*)(ws + 3 * seg);

  dim3 ggrid(B_ * S_ / 16, D_ / 64);  // (512, 16), 1 wave per 16x16 tile
  dim3 gblk(128);

  gemm16<0><<<ggrid, gblk, 0, stream>>>(q_in, nullptr, wq, bq, Qb, nullptr);
  gemm16<1><<<ggrid, gblk, 0, stream>>>(k_in, nullptr, wk, bk, Kb, nullptr);
  gemm16<2><<<ggrid, gblk, 0, stream>>>(v_in, nullptr, wv, bv, Vt, nullptr);

  dim3 agrid(S_ / 128, B_ * H_);  // (16, 64)
  attn_kernel<<<agrid, gblk, 0, stream>>>(Qb, Kb, Vt, Att);

  gemm16<3><<<ggrid, gblk, 0, stream>>>(nullptr, Att, wo, bo, nullptr, out);
}
